// QAKT_979252544348
// MI455X (gfx1250) — matrix operand order repaired and checked by reading, not compiled, not run on hardware
//
#include <hip/hip_runtime.h>
#include <hip/hip_bf16.h>
#include <math.h>

// ---------------------------------------------------------------------------
// AKT forward for MI455X (gfx1250, wave32, WMMA 16x16x32 f16->f32).
// All dense GEMMs run through v_wmma_f32_16x16x32_f16; attention computes
// Q*K^T and P*V with WMMA, with the AKT distance-decay double-softmax done
// per-row in LDS (16 x 1024 fp32 tile = 64KB < 320KB/WGP).
// ---------------------------------------------------------------------------

typedef __attribute__((ext_vector_type(16))) _Float16 v16h;
typedef __attribute__((ext_vector_type(8)))  float    v8f;

#define BQ   8
#define SQ   1024
#define BSQ  8192          // B*S
#define DQ   256
#define D2Q  512
#define DFFQ 2048
#define HQ   8
#define NQ1Q 1025
#define KPQ  1056          // 1025 padded up to multiple of 32
#define NPIDQ 4096
#define M3Q  24576         // 3*B*S gathered embedding rows

// ---------------- WMMA fragment helpers (contiguous k = 16*(lane>>4) + e, same for A and B) ----------------

__device__ __forceinline__ v8f wmma32(const v16h a, const v16h b, const v8f c) {
  return __builtin_amdgcn_wmma_f32_16x16x32_f16(
      /*neg_a=*/false, a, /*neg_b=*/false, b,
      /*c_mod=*/(short)0, c, /*reuse_a=*/false, /*reuse_b=*/false);
}

// A fragment, 16x32 f16, row-major source with leading dim lda.
// lane L: m = L&15, k = 16*(L>>4) + e
__device__ __forceinline__ v16h load_a_frag(const _Float16* A, int lda, int lane) {
  const int m  = lane & 15;
  const int kb = (lane >> 4) * 16;
  const _Float16* p = A + (long)m * lda + kb;
  v16h f;
#pragma unroll
  for (int e = 0; e < 8; ++e) f[e] = p[e];
#pragma unroll
  for (int e = 0; e < 8; ++e) f[8 + e] = p[8 + e];
  return f;
}

// A fragment from an LDS fp32 row tile (P matrix), converting to f16.
__device__ __forceinline__ v16h load_a_frag_lds(const float* sc, int ldc, int k0, int lane) {
  const int m  = lane & 15;
  const int kb = (lane >> 4) * 16;
  const float* p = sc + m * ldc + k0 + kb;
  v16h f;
#pragma unroll
  for (int e = 0; e < 8; ++e) f[e] = (_Float16)p[e];
#pragma unroll
  for (int e = 0; e < 8; ++e) f[8 + e] = (_Float16)p[8 + e];
  return f;
}

// B fragment, 32x16 f16, B[k][n] row-major with leading dim ldb.
// lane L: n = L&15, k = 16*(L>>4) + e
__device__ __forceinline__ v16h load_b_frag(const _Float16* B, int ldb, int lane) {
  const int n  = lane & 15;
  const int kb = (lane >> 4) * 16;
  const _Float16* p = B + (long)kb * ldb + n;
  v16h f;
#pragma unroll
  for (int e = 0; e < 16; ++e) f[e] = p[(long)e * ldb];
  return f;
}

// B fragment for K^T: B[k][n] = Kmat[n][k] (Kmat row-major, leading dim ld).
// Contiguous 16-half read per lane.
__device__ __forceinline__ v16h load_bT_frag(const _Float16* Km, int ld, int lane) {
  const int n  = lane & 15;
  const int kb = (lane >> 4) * 16;
  const _Float16* p = Km + (long)n * ld + kb;
  v16h f;
#pragma unroll
  for (int e = 0; e < 16; ++e) f[e] = p[e];
  return f;
}

// ---------------- generic GEMM: C = act(A[MxK]f16 @ B[KxN]f16 + bias) -------
// block = 256 threads = 8 waves; block tile 128(M) x 64(N); wave w -> 16 rows.
template <int ACT>  // 0 = none, 1 = relu
__global__ void gemm_wmma(const _Float16* __restrict__ A,
                          const _Float16* __restrict__ Bm,
                          const float* __restrict__ bias,
                          float* __restrict__ C, int M, int N, int K) {
  const int wave = threadIdx.x >> 5;
  const int lane = threadIdx.x & 31;
  const int row0 = blockIdx.x * 128 + wave * 16;
  const int col0 = blockIdx.y * 64;

  v8f acc[4];
#pragma unroll
  for (int t = 0; t < 4; ++t)
#pragma unroll
    for (int r = 0; r < 8; ++r) acc[t][r] = 0.0f;

  for (int k0 = 0; k0 < K; k0 += 32) {
    const v16h a = load_a_frag(A + (long)row0 * K + k0, K, lane);
#pragma unroll
    for (int t = 0; t < 4; ++t) {
      const v16h bfrag = load_b_frag(Bm + (long)k0 * N + col0 + t * 16, N, lane);
      acc[t] = wmma32(a, bfrag, acc[t]);
    }
  }

  const int n  = lane & 15;
  const int mb = (lane >> 4) * 8;
#pragma unroll
  for (int t = 0; t < 4; ++t) {
    const int cc = col0 + t * 16 + n;
    const float bv = bias[cc];
#pragma unroll
    for (int r = 0; r < 8; ++r) {
      float v = acc[t][r] + bv;
      if (ACT == 1) v = fmaxf(v, 0.0f);
      C[(long)(row0 + mb + r) * N + cc] = v;
    }
  }
}

// ---------------- AKT attention ---------------------------------------------
// One wave per (q-tile of 16, b*H+h). Q doubles as K (kq_same in reference).
__global__ void __launch_bounds__(32)
akt_attn(const _Float16* __restrict__ Qf,  // [BS, dmq] f16, head h at cols h*dhq
         const _Float16* __restrict__ Vf,  // [BS, dmv] f16, head h at cols h*dhv
         float* __restrict__ O,            // [BS, dmv] f32
         const float* __restrict__ gam,    // [H]
         int dmq, int dhq, int dmv, int dhv, int mask_k, int zero_pad) {
  __shared__ float sc[16 * SQ];  // 64 KB score tile
  const int lane = threadIdx.x;
  const int qt = blockIdx.x;           // 0..S/16-1
  const int bh = blockIdx.y;           // 0..B*H-1
  const int b = bh / HQ, h = bh % HQ;
  const int qrow = b * SQ + qt * 16;   // global row of first query
  const float scale = rsqrtf((float)dhq);

  // -------- scores = (Q K^T) * scale ----------
  for (int kt = 0; kt < SQ / 16; ++kt) {
    v8f acc;
#pragma unroll
    for (int r = 0; r < 8; ++r) acc[r] = 0.0f;
    for (int ks = 0; ks < dhq; ks += 32) {
      const v16h a  = load_a_frag(Qf + (long)qrow * dmq + h * dhq + ks, dmq, lane);
      const v16h bt = load_bT_frag(Qf + (long)(b * SQ + kt * 16) * dmq + h * dhq + ks, dmq, lane);
      acc = wmma32(a, bt, acc);
    }
    const int n = lane & 15, mb = (lane >> 4) * 8;
#pragma unroll
    for (int r = 0; r < 8; ++r)
      sc[(mb + r) * SQ + kt * 16 + n] = acc[r] * scale;
  }
  __syncthreads();

  // -------- AKT double softmax with distance decay (lanes 0..15, 1 row each)
  if (lane < 16) {
    const int qi = qt * 16 + lane;              // sequence position of query
    float* row = sc + lane * SQ;
    const int limit = mask_k ? qi : qi - 1;     // inclusive max allowed key
    const float gamma = -log1pf(__expf(gam[h]));  // -softplus(gam)

    // pass 1: masked max
    float mx = -1e32f;
    for (int j = 0; j <= limit; ++j) mx = fmaxf(mx, row[j]);
    // pass 2: masked exp-sum
    float sumE = 0.0f;
    for (int j = 0; j <= limit; ++j) sumE += __expf(row[j] - mx);
    const float inv = (sumE > 0.0f) ? 1.0f / sumE : 0.0f;
    const float disttot = (limit >= 0) ? 1.0f : 0.0f;

    // pass 3: cumsum -> distance effect -> rescored logits v2, track max2
    float cum = 0.0f, mx2 = -1e32f;
    for (int j = 0; j < SQ; ++j) {
      float v2;
      if (j <= limit) {
        const float e = __expf(row[j] - mx);
        cum += e;
        const float pc = cum * inv;                       // cumsum of s_
        const float t = (disttot - pc) * fabsf((float)(j - qi));
        const float dist = sqrtf(fmaxf(t, 0.0f));
        float eff = __expf(gamma * dist);
        eff = fminf(fmaxf(eff, 1e-5f), 1e5f);
        v2 = row[j] * eff;
      } else {
        v2 = -1e32f;
      }
      row[j] = v2;
      mx2 = fmaxf(mx2, v2);
    }
    // pass 4: softmax #2 denominator over the full row
    float s2 = 0.0f;
    for (int j = 0; j < SQ; ++j) s2 += __expf(row[j] - mx2);
    const float i2 = 1.0f / s2;
    const bool zp = (zero_pad != 0) && (qi == 0);
    for (int j = 0; j < SQ; ++j)
      row[j] = zp ? 0.0f : __expf(row[j] - mx2) * i2;
  }
  __syncthreads();

  // -------- O = P @ V ----------
  for (int nt = 0; nt < dhv; nt += 16) {
    v8f acc;
#pragma unroll
    for (int r = 0; r < 8; ++r) acc[r] = 0.0f;
    for (int k0 = 0; k0 < SQ; k0 += 32) {
      const v16h a  = load_a_frag_lds(sc, SQ, k0, lane);
      const v16h vb = load_b_frag(Vf + (long)(b * SQ + k0) * dmv + h * dhv + nt, dmv, lane);
      acc = wmma32(a, vb, acc);
    }
    const int n = lane & 15, mb = (lane >> 4) * 8;
#pragma unroll
    for (int r = 0; r < 8; ++r)
      O[(long)(qrow + mb + r) * dmv + h * dhv + nt + n] = acc[r];
  }
}

// ---------------- small elementwise / reduction kernels ---------------------

__global__ void cast_f16_k(const float* __restrict__ s, _Float16* __restrict__ d, long n) {
  long i = (long)blockIdx.x * blockDim.x + threadIdx.x;
  const long stride = (long)gridDim.x * blockDim.x;
  for (; i < n; i += stride) d[i] = (_Float16)s[i];
}

__global__ void cast_pad_k(const float* __restrict__ s, _Float16* __restrict__ d,
                           int rows, int cols, int prows) {
  long i = (long)blockIdx.x * blockDim.x + threadIdx.x;
  const long n = (long)prows * cols;
  const long stride = (long)gridDim.x * blockDim.x;
  for (; i < n; i += stride) {
    const int r = (int)(i / cols), c = (int)(i % cols);
    d[i] = (r < rows) ? (_Float16)s[(long)r * cols + c] : (_Float16)0.0f;
  }
}

// gather sigmoid(pemb_W[idx]) rows into padded f16 A matrix + per-row sums
__global__ void gather_embed(const int* __restrict__ pdat, const int* __restrict__ padat,
                             const float* __restrict__ pembW,
                             _Float16* __restrict__ A, float* __restrict__ rows) {
  const int r = blockIdx.x;  // 0..M3Q-1
  const int tid = threadIdx.x;
  const int region = r / BSQ;
  const int t = r - region * BSQ;
  int idx;
  if (region == 0) {
    idx = pdat[t];
  } else {
    const int pa = padat[t];
    idx = (region == 1) ? (pa > NPIDQ ? pa - NPIDQ : 0)
                        : (pa <= NPIDQ ? pa : 0);
  }
  const float* src = pembW + (long)idx * NQ1Q;
  _Float16* dst = A + (long)r * KPQ;
  float s = 0.0f;
  for (int c = tid; c < KPQ; c += 256) {
    float v = 0.0f;
    if (c < NQ1Q) v = 1.0f / (1.0f + __expf(-src[c]));
    dst[c] = (_Float16)v;
    s += v;
  }
  __shared__ float red[256];
  red[tid] = s;
  __syncthreads();
  for (int o = 128; o > 0; o >>= 1) {
    if (tid < o) red[tid] += red[tid + o];
    __syncthreads();
  }
  if (tid == 0) rows[r] = red[0];
}

__global__ void finish_q(const float* __restrict__ E, const float* __restrict__ rows,
                         const int* __restrict__ pdat, const float* __restrict__ diffW,
                         float* __restrict__ q_embed) {
  const long i = (long)blockIdx.x * 256 + threadIdx.x;  // BS*256 threads
  if (i >= (long)BSQ * DQ) return;
  const int t = (int)(i >> 8);
  const float pid = diffW[pdat[t]];
  q_embed[i] = E[i] / rows[t] + pid;
}

__global__ void finish_qa(const float* __restrict__ E, const float* __restrict__ rows,
                          const int* __restrict__ pdat, const float* __restrict__ diffW,
                          float* __restrict__ qa) {
  const long i = (long)blockIdx.x * 256 + threadIdx.x;  // BS*512 threads
  if (i >= (long)BSQ * D2Q) return;
  const int t = (int)(i >> 9);
  const int c = (int)(i & 511);
  const float pid = diffW[pdat[t]];
  float v;
  if (c < 256) v = E[(long)(BSQ + t) * DQ + c] / rows[BSQ + t];
  else         v = E[(long)(2 * BSQ + t) * DQ + (c - 256)] / rows[2 * BSQ + t];
  qa[i] = v + pid;
}

// out = LN(x (+res)) * g + b, one block per row
__global__ void ln_kernel(const float* __restrict__ x, const float* __restrict__ res,
                          const float* __restrict__ g, const float* __restrict__ bta,
                          float* __restrict__ out, int dim) {
  const int r = blockIdx.x;
  const int tid = threadIdx.x;
  const float* xr = x + (long)r * dim;
  const float* rr = res ? res + (long)r * dim : nullptr;
  float vloc[2];
  int cnt = 0;
  float s = 0.0f;
  for (int c = tid; c < dim; c += 256) {
    const float v = xr[c] + (rr ? rr[c] : 0.0f);
    vloc[cnt++] = v;
    s += v;
  }
  __shared__ float red[256];
  red[tid] = s;
  __syncthreads();
  for (int o = 128; o > 0; o >>= 1) { if (tid < o) red[tid] += red[tid + o]; __syncthreads(); }
  const float mean = red[0] / dim;
  __syncthreads();
  s = 0.0f;
  cnt = 0;
  for (int c = tid; c < dim; c += 256) { const float d = vloc[cnt++] - mean; s += d * d; }
  red[tid] = s;
  __syncthreads();
  for (int o = 128; o > 0; o >>= 1) { if (tid < o) red[tid] += red[tid + o]; __syncthreads(); }
  const float rs = rsqrtf(red[0] / dim + 1e-5f);
  cnt = 0;
  for (int c = tid; c < dim; c += 256)
    out[(long)r * dim + c] = g[c] * (vloc[cnt++] - mean) * rs + bta[c];
}

__global__ void concat_k(const float* __restrict__ a, const float* __restrict__ b,
                         float* __restrict__ c) {
  const long i = (long)blockIdx.x * 256 + threadIdx.x;  // BS*512
  if (i >= (long)BSQ * D2Q) return;
  const int t = (int)(i >> 9);
  const int col = (int)(i & 511);
  c[i] = (col < 256) ? a[(long)t * DQ + col] : b[(long)t * DQ + (col - 256)];
}

__global__ void gemv_out(const float* __restrict__ X, const float* __restrict__ w,
                         const float* __restrict__ b, float* __restrict__ out) {
  const int r = blockIdx.x;
  const int tid = threadIdx.x;  // 256 = DQ
  __shared__ float red[256];
  red[tid] = X[(long)r * DQ + tid] * w[tid];
  __syncthreads();
  for (int o = 128; o > 0; o >>= 1) { if (tid < o) red[tid] += red[tid + o]; __syncthreads(); }
  if (tid == 0) out[r] = red[0] + b[0];
}

__global__ void zero1_k(float* p) { *p = 0.0f; }

__global__ void creg_k(const int* __restrict__ pdat, const float* __restrict__ diffW,
                       float* __restrict__ dst) {
  const int tid = threadIdx.x;
  long i = (long)blockIdx.x * 256 + tid;
  float s = 0.0f;
  for (; i < BSQ; i += (long)gridDim.x * 256) {
    const float d = diffW[pdat[i]];
    s += d * d;
  }
  __shared__ float red[256];
  red[tid] = s;
  __syncthreads();
  for (int o = 128; o > 0; o >>= 1) { if (tid < o) red[tid] += red[tid + o]; __syncthreads(); }
  if (tid == 0) atomicAdd(dst, red[0] * 1e-5f);
}

// ---------------------------------------------------------------------------

static inline void cast16(const float* s, _Float16* d, long n, hipStream_t st) {
  const int blocks = (int)(((n + 255) / 256) > 4096 ? 4096 : ((n + 255) / 256));
  cast_f16_k<<<blocks, 256, 0, st>>>(s, d, n);
}

extern "C" void kernel_launch(void* const* d_in, const int* in_sizes, int n_in,
                              void* d_out, int out_size, void* d_ws, size_t ws_size,
                              hipStream_t stream) {
  (void)in_sizes; (void)n_in; (void)out_size; (void)ws_size;
  // ---- inputs (setup_inputs order) ----
  const int*   p_data = (const int*)d_in[0];
  const int*   pa_data= (const int*)d_in[1];
  const float* pemb_W = (const float*)d_in[3];
  const float* diff_W = (const float*)d_in[4];
  const float* qemb_W = (const float*)d_in[5];
  const float* qemb_b = (const float*)d_in[6];
  const float* y_kW = (const float*)d_in[7];  const float* y_kb = (const float*)d_in[8];
  const float* y_vW = (const float*)d_in[9];  const float* y_vb = (const float*)d_in[10];
  const float* y_oW = (const float*)d_in[11]; const float* y_ob = (const float*)d_in[12];
  const float* y_gam = (const float*)d_in[13];
  const float* y_ln1g = (const float*)d_in[14]; const float* y_ln1b = (const float*)d_in[15];
  const float* y_f1W = (const float*)d_in[16]; const float* y_f1b = (const float*)d_in[17];
  const float* y_f2W = (const float*)d_in[18]; const float* y_f2b = (const float*)d_in[19];
  const float* y_ln2g = (const float*)d_in[20]; const float* y_ln2b = (const float*)d_in[21];
  const float* x_kW = (const float*)d_in[22]; const float* x_kb = (const float*)d_in[23];
  const float* x_vW = (const float*)d_in[24]; const float* x_vb = (const float*)d_in[25];
  const float* x_oW = (const float*)d_in[26]; const float* x_ob = (const float*)d_in[27];
  const float* x_gam = (const float*)d_in[28];
  const float* x_ln1g = (const float*)d_in[29]; const float* x_ln1b = (const float*)d_in[30];
  const float* h_kW = (const float*)d_in[31]; const float* h_kb = (const float*)d_in[32];
  const float* h_vW = (const float*)d_in[33]; const float* h_vb = (const float*)d_in[34];
  const float* h_oW = (const float*)d_in[35]; const float* h_ob = (const float*)d_in[36];
  const float* h_gam = (const float*)d_in[37];
  const float* h_ln1g = (const float*)d_in[38]; const float* h_ln1b = (const float*)d_in[39];
  const float* h_f1W = (const float*)d_in[40]; const float* h_f1b = (const float*)d_in[41];
  const float* h_f2W = (const float*)d_in[42]; const float* h_f2b = (const float*)d_in[43];
  const float* h_ln2g = (const float*)d_in[44]; const float* h_ln2b = (const float*)d_in[45];
  const float* oln1g = (const float*)d_in[46]; const float* oln1b = (const float*)d_in[47];
  const float* o1W = (const float*)d_in[48]; const float* o1b = (const float*)d_in[49];
  const float* oln2g = (const float*)d_in[50]; const float* oln2b = (const float*)d_in[51];
  const float* o2W = (const float*)d_in[52]; const float* o2b = (const float*)d_in[53];
  const float* oln3g = (const float*)d_in[54]; const float* oln3b = (const float*)d_in[55];
  const float* o3W = (const float*)d_in[56]; const float* o3b = (const float*)d_in[57];
  float* out = (float*)d_out;

  // ---- workspace bump allocator ----
  char* ws = (char*)d_ws;
  size_t off = 0;
  auto alloc = [&](size_t bytes) -> char* {
    off = (off + 255) & ~(size_t)255;
    char* p = ws + off;
    off += bytes;
    return p;
  };
  _Float16* AEMB = (_Float16*)alloc((size_t)M3Q * KPQ * 2);
  float*    ROWS = (float*)   alloc((size_t)M3Q * 4);
  _Float16* WqembW = (_Float16*)alloc((size_t)KPQ * DQ * 2);
  _Float16* WykW  = (_Float16*)alloc((size_t)D2Q * D2Q * 2);
  _Float16* WyvW  = (_Float16*)alloc((size_t)D2Q * D2Q * 2);
  _Float16* WyoW  = (_Float16*)alloc((size_t)D2Q * D2Q * 2);
  _Float16* Wyf1  = (_Float16*)alloc((size_t)D2Q * DFFQ * 2);
  _Float16* Wyf2  = (_Float16*)alloc((size_t)DFFQ * D2Q * 2);
  _Float16* WxkW  = (_Float16*)alloc((size_t)DQ * DQ * 2);
  _Float16* WxvW  = (_Float16*)alloc((size_t)DQ * DQ * 2);
  _Float16* WxoW  = (_Float16*)alloc((size_t)DQ * DQ * 2);
  _Float16* WhkW  = (_Float16*)alloc((size_t)DQ * DQ * 2);
  _Float16* WhvW  = (_Float16*)alloc((size_t)D2Q * D2Q * 2);
  _Float16* WhoW  = (_Float16*)alloc((size_t)D2Q * DQ * 2);
  _Float16* Whf1  = (_Float16*)alloc((size_t)DQ * DFFQ * 2);
  _Float16* Whf2  = (_Float16*)alloc((size_t)DFFQ * DQ * 2);
  _Float16* Wo1   = (_Float16*)alloc((size_t)D2Q * D2Q * 2);
  _Float16* Wo2   = (_Float16*)alloc((size_t)D2Q * DQ * 2);
  float*    T0   = (float*)alloc((size_t)BSQ * DFFQ * 4);   // 64 MB
  float*    T1   = (float*)alloc((size_t)BSQ * D2Q * 4);
  float*    T2   = (float*)alloc((size_t)BSQ * D2Q * 4);
  _Float16* BF0  = (_Float16*)alloc((size_t)BSQ * DFFQ * 2); // 32 MB
  _Float16* BF1  = (_Float16*)alloc((size_t)BSQ * D2Q * 2);
  float*    QEMB = (float*)alloc((size_t)BSQ * DQ * 4);
  float*    QAEMB= (float*)alloc((size_t)BSQ * D2Q * 4);
  float*    YBUF = (float*)alloc((size_t)BSQ * D2Q * 4);
  float*    X1   = (float*)alloc((size_t)BSQ * DQ * 4);
  float*    XF   = (float*)alloc((size_t)BSQ * DQ * 4);

  // ---- cast weights to f16 ----
  cast_pad_k<<<4096, 256, 0, stream>>>(qemb_W, WqembW, NQ1Q, DQ, KPQ);
  cast16(y_kW, WykW, (long)D2Q * D2Q, stream);
  cast16(y_vW, WyvW, (long)D2Q * D2Q, stream);
  cast16(y_oW, WyoW, (long)D2Q * D2Q, stream);
  cast16(y_f1W, Wyf1, (long)D2Q * DFFQ, stream);
  cast16(y_f2W, Wyf2, (long)DFFQ * D2Q, stream);
  cast16(x_kW, WxkW, (long)DQ * DQ, stream);
  cast16(x_vW, WxvW, (long)DQ * DQ, stream);
  cast16(x_oW, WxoW, (long)DQ * DQ, stream);
  cast16(h_kW, WhkW, (long)DQ * DQ, stream);
  cast16(h_vW, WhvW, (long)D2Q * D2Q, stream);
  cast16(h_oW, WhoW, (long)D2Q * DQ, stream);
  cast16(h_f1W, Whf1, (long)DQ * DFFQ, stream);
  cast16(h_f2W, Whf2, (long)DFFQ * DQ, stream);
  cast16(o1W, Wo1, (long)D2Q * D2Q, stream);
  cast16(o2W, Wo2, (long)D2Q * DQ, stream);

  // ---- embeddings ----
  gather_embed<<<M3Q, 256, 0, stream>>>(p_data, pa_data, pemb_W, AEMB, ROWS);
  gemm_wmma<1><<<dim3(M3Q / 128, DQ / 64), 256, 0, stream>>>(AEMB, WqembW, qemb_b, T0, M3Q, DQ, KPQ);
  finish_q<<<(BSQ * DQ) / 256, 256, 0, stream>>>(T0, ROWS, p_data, diff_W, QEMB);
  finish_qa<<<(BSQ * D2Q) / 256, 256, 0, stream>>>(T0, ROWS, p_data, diff_W, QAEMB);

  const dim3 attnGrid(SQ / 16, BQ * HQ);

  // ---- y layer (dm=512, dh=64, mask_k=1, zero_pad=0, FF) ----
  cast16(QAEMB, BF1, (long)BSQ * D2Q, stream);
  gemm_wmma<0><<<dim3(BSQ / 128, D2Q / 64), 256, 0, stream>>>(BF1, WykW, y_kb, T1, BSQ, D2Q, D2Q);
  gemm_wmma<0><<<dim3(BSQ / 128, D2Q / 64), 256, 0, stream>>>(BF1, WyvW, y_vb, T2, BSQ, D2Q, D2Q);
  cast16(T1, BF1, (long)BSQ * D2Q, stream);   // Q==K
  cast16(T2, BF0, (long)BSQ * D2Q, stream);   // V
  akt_attn<<<attnGrid, 32, 0, stream>>>(BF1, BF0, T0, y_gam, D2Q, 64, D2Q, 64, 1, 0);
  cast16(T0, BF1, (long)BSQ * D2Q, stream);
  gemm_wmma<0><<<dim3(BSQ / 128, D2Q / 64), 256, 0, stream>>>(BF1, WyoW, y_ob, T1, BSQ, D2Q, D2Q);
  ln_kernel<<<BSQ, 256, 0, stream>>>(T1, QAEMB, y_ln1g, y_ln1b, T2, D2Q);
  cast16(T2, BF1, (long)BSQ * D2Q, stream);
  gemm_wmma<1><<<dim3(BSQ / 128, DFFQ / 64), 256, 0, stream>>>(BF1, Wyf1, y_f1b, T0, BSQ, DFFQ, D2Q);
  cast16(T0, BF0, (long)BSQ * DFFQ, stream);
  gemm_wmma<0><<<dim3(BSQ / 128, D2Q / 64), 256, 0, stream>>>(BF0, Wyf2, y_f2b, T1, BSQ, D2Q, DFFQ);
  ln_kernel<<<BSQ, 256, 0, stream>>>(T1, T2, y_ln2g, y_ln2b, YBUF, D2Q);

  // ---- x layer (dm=256, dh=32, mask_k=1, zero_pad=0, no FF) ----
  cast16(QEMB, BF1, (long)BSQ * DQ, stream);
  gemm_wmma<0><<<dim3(BSQ / 128, DQ / 64), 256, 0, stream>>>(BF1, WxkW, x_kb, T1, BSQ, DQ, DQ);
  gemm_wmma<0><<<dim3(BSQ / 128, DQ / 64), 256, 0, stream>>>(BF1, WxvW, x_vb, T2, BSQ, DQ, DQ);
  cast16(T1, BF1, (long)BSQ * DQ, stream);
  cast16(T2, BF0, (long)BSQ * DQ, stream);
  akt_attn<<<attnGrid, 32, 0, stream>>>(BF1, BF0, T0, x_gam, DQ, 32, DQ, 32, 1, 0);
  cast16(T0, BF1, (long)BSQ * DQ, stream);
  gemm_wmma<0><<<dim3(BSQ / 128, DQ / 64), 256, 0, stream>>>(BF1, WxoW, x_ob, T1, BSQ, DQ, DQ);
  ln_kernel<<<BSQ, 256, 0, stream>>>(T1, QEMB, x_ln1g, x_ln1b, X1, DQ);

  // ---- h layer (q=k from x1 dh=32, v from y dh=64, mask_k=0, zero_pad=1, FF)
  cast16(X1, BF1, (long)BSQ * DQ, stream);
  gemm_wmma<0><<<dim3(BSQ / 128, DQ / 64), 256, 0, stream>>>(BF1, WhkW, h_kb, T1, BSQ, DQ, DQ);
  cast16(YBUF, BF1, (long)BSQ * D2Q, stream);
  gemm_wmma<0><<<dim3(BSQ / 128, D2Q / 64), 256, 0, stream>>>(BF1, WhvW, h_vb, T2, BSQ, D2Q, D2Q);
  cast16(T1, BF1, (long)BSQ * DQ, stream);    // Qh == Kh
  cast16(T2, BF0, (long)BSQ * D2Q, stream);   // Vh
  akt_attn<<<attnGrid, 32, 0, stream>>>(BF1, BF0, T0, h_gam, DQ, 32, D2Q, 64, 0, 1);
  cast16(T0, BF1, (long)BSQ * D2Q, stream);
  gemm_wmma<0><<<dim3(BSQ / 128, DQ / 64), 256, 0, stream>>>(BF1, WhoW, h_ob, T1, BSQ, DQ, D2Q);
  ln_kernel<<<BSQ, 256, 0, stream>>>(T1, X1, h_ln1g, h_ln1b, T2, DQ);   // x2 in T2
  cast16(T2, BF1, (long)BSQ * DQ, stream);
  gemm_wmma<1><<<dim3(BSQ / 128, DFFQ / 64), 256, 0, stream>>>(BF1, Whf1, h_f1b, T0, BSQ, DFFQ, DQ);
  cast16(T0, BF0, (long)BSQ * DFFQ, stream);
  gemm_wmma<0><<<dim3(BSQ / 128, DQ / 64), 256, 0, stream>>>(BF0, Whf2, h_f2b, T1, BSQ, DQ, DFFQ);
  ln_kernel<<<BSQ, 256, 0, stream>>>(T1, T2, h_ln2g, h_ln2b, XF, DQ);

  // ---- output head ----
  concat_k<<<(BSQ * D2Q) / 256, 256, 0, stream>>>(XF, QEMB, T1);
  ln_kernel<<<BSQ, 256, 0, stream>>>(T1, nullptr, oln1g, oln1b, T2, D2Q);
  cast16(T2, BF1, (long)BSQ * D2Q, stream);
  gemm_wmma<1><<<dim3(BSQ / 128, D2Q / 64), 256, 0, stream>>>(BF1, Wo1, o1b, T0, BSQ, D2Q, D2Q);
  ln_kernel<<<BSQ, 256, 0, stream>>>(T0, nullptr, oln2g, oln2b, T2, D2Q);
  cast16(T2, BF1, (long)BSQ * D2Q, stream);
  gemm_wmma<1><<<dim3(BSQ / 128, DQ / 64), 256, 0, stream>>>(BF1, Wo2, o2b, T0, BSQ, DQ, D2Q);
  ln_kernel<<<BSQ, 256, 0, stream>>>(T0, nullptr, oln3g, oln3b, T2, DQ);
  gemv_out<<<BSQ, 256, 0, stream>>>(T2, o3W, o3b, out);

  // ---- c_reg ----
  zero1_k<<<1, 1, 0, stream>>>(out + BSQ);
  creg_k<<<32, 256, 0, stream>>>(p_data, diff_W, out + BSQ);
}
